// GraphAttentionLayer_72877005078913
// MI455X (gfx1250) — matrix operand order repaired and checked by reading, not compiled, not run on hardware
//
#include <hip/hip_runtime.h>
#include <cmath>

typedef __attribute__((ext_vector_type(16))) __bf16 v16bf;
typedef __attribute__((ext_vector_type(8)))  __bf16 v8bf;
typedef __attribute__((ext_vector_type(8)))  float  v8f;

#define NROW 8192
#define FIN  256
#define FOUT 64
#define NW   256          // 32-bit mask words per adjacency row
#define ALPHA 0.2f

// workspace byte offsets (total ~10.95 MiB)
#define WHBT_OFF  (0u)          // bf16 Wh^T  [64][8192]   1.00 MiB
#define FSRC_OFF  (0x100000u)   // f32 [8192]
#define FDST_OFF  (0x108000u)   // f32 [8192]
#define MMAX_OFF  (0x110000u)   // f32 [1]
#define MASK_OFF  (0x120000u)   // u32 [8192][256]         8.00 MiB (L2-resident)
#define TBT_OFF   (0x920000u)   // bf16 t^T   [80][8192]   1.25 MiB (ones col appended)
#define E1C_OFF   (0xA60000u)   // f32 exp(f_dst)          32 KB
#define E2C_OFF   (0xA68000u)   // f32 exp(0.2*f_dst)      32 KB

// ---------------------------------------------------------------------------
// Tensor Data Mover: 2D tile load of `rows` x 256 u32 words into LDS.
// D# per CDNA5 ISA sec 8.3/8.4: count=1, type=2(image), data_size=4B,
// tensor_dim0 = tile_dim0 = tensor_dim0_stride = 256 words, tile_dim1 = rows.
// Arity hedge: clang-22 (ROCm 7.2) = 5 args, clang-23 = 6 args.
// ---------------------------------------------------------------------------
#if __has_builtin(__builtin_amdgcn_tensor_load_to_lds) && \
    __has_builtin(__builtin_amdgcn_s_wait_tensorcnt)
#define HAVE_TDM 1
typedef __attribute__((ext_vector_type(4))) unsigned int v4u;
typedef __attribute__((ext_vector_type(8))) int          v8i;
typedef __attribute__((ext_vector_type(4))) int          v4i;

__device__ __forceinline__ void tdm_load_mask(const unsigned int* gsrc,
                                              unsigned int* lds_dst,
                                              unsigned int rows)
{
    const unsigned long long ga = (unsigned long long)gsrc;
    const unsigned int lds_addr = (unsigned int)(unsigned long long)lds_dst;
    v4u g0;
    g0[0] = 1u;                                                // count=1
    g0[1] = lds_addr;                                          // lds_addr
    g0[2] = (unsigned int)(ga & 0xffffffffu);                  // global_addr lo
    g0[3] = (unsigned int)((ga >> 32) & 0x01ffffffu) | (2u << 30); // hi + type=2
    v8i g1;
    g1[0] = (int)(2u << 16);          // wg_mask=0, data_size=2 (4 bytes)
    g1[1] = (int)(256u << 16);        // tensor_dim0 = 256 (bits 79:48 lo half)
    g1[2] = (int)(rows << 16);        // tensor_dim0 hi=0 | tensor_dim1 lo = rows
    g1[3] = (int)(256u << 16);        // tensor_dim1 hi=0 | tile_dim0 = 256
    g1[4] = (int)rows;                // tile_dim1 = rows, tile_dim2 = 0
    g1[5] = 256;                      // tensor_dim0_stride lo
    g1[6] = 0;                        // stride hi | tensor_dim1_stride lo
    g1[7] = 0;
    const v4i z4 = {0, 0, 0, 0};
#if defined(__clang_major__) && (__clang_major__ >= 23)
    const v8i z8 = {0, 0, 0, 0, 0, 0, 0, 0};
    __builtin_amdgcn_tensor_load_to_lds(g0, g1, z4, z4, z8, 0);
#else
    __builtin_amdgcn_tensor_load_to_lds(g0, g1, z4, z4, 0);
#endif
}
#endif

// ---------------------------------------------------------------------------
// k_t A-tile build: 4 adjacency bits -> two packed-bf16 words via 16-entry
// LDS LUT (bf16 1.0 = 0x3F80). 8 elements/thread: 2 ds_load_b64 + 1 store.
// Tile: 32 rows x 32 K bf16, row-major, row stride 32.
// ---------------------------------------------------------------------------
__device__ __forceinline__ void build_adj_tile(const unsigned int* smask,
                                               const uint2* slut,
                                               unsigned int* dst, int chunk,
                                               int tid)
{
    const int r  = tid >> 2;          // 0..31 (128 threads)
    const int kq = (tid & 3) * 8;     // K sub-block 0,8,16,24
    const unsigned int w = smask[r * NW + chunk] >> kq;
    const uint2 a = slut[w & 15u];
    const uint2 b = slut[(w >> 4) & 15u];
    uint4 v = {a.x, a.y, b.x, b.y};
    *(uint4*)&dst[r * 16 + (kq >> 1)] = v;    // 16B-aligned
}

// ---------------------------------------------------------------------------
// k_attn P-tile build, exp-free inner loop via LeakyReLU separability:
//   e>0 : exp(e-mi) = exp(fs-mi)        * exp(fd)      = R1[i]*E1c[j]
//   e<=0: exp(.2e-mi) = exp(.2fs-mi)    * exp(.2fd)    = R2[i]*E2c[j]
//   branch: fs+fd>0  <=>  E1c[j] > exp(-fs) = Tr[i]
// ---------------------------------------------------------------------------
__device__ __forceinline__ void build_p_tile(const unsigned int* smask,
                                             const float* sE1, const float* sE2,
                                             const float* sR1, const float* sR2,
                                             const float* sTr, unsigned int* dst,
                                             int chunk, int tid)
{
    for (int p = tid; p < 512; p += 160) {    // 512 bf16 pairs, 160 threads
        const int r  = p >> 4;                // row 0..31
        const int pi = p & 15;                // pair 0..15 (K = 2*pi, 2*pi+1)
        const unsigned int w = smask[r * NW + chunk] >> (pi * 2);
        const int j0 = chunk * 32 + pi * 2;
        const float2 c1 = *(const float2*)&sE1[j0];
        const float2 c2 = *(const float2*)&sE2[j0];
        const float tr = sTr[r], r1 = sR1[r], r2 = sR2[r];
        float p0 = (c1.x > tr) ? r1 * c1.x : r2 * c2.x;
        float p1 = (c1.y > tr) ? r1 * c1.y : r2 * c2.y;
        p0 = (w & 1u) ? p0 : 0.f;             // adjacency mask
        p1 = (w & 2u) ? p1 : 0.f;
        const unsigned short u0 = __builtin_bit_cast(unsigned short, (__bf16)p0);
        const unsigned short u1 = __builtin_bit_cast(unsigned short, (__bf16)p1);
        dst[r * 16 + pi] = ((unsigned int)u1 << 16) | u0;
    }
}

// A-operand fetch from a 32x32 LDS tile in contiguous k order (not the ISA table order):
// lane m = lane&15, slots e0-15 -> K = half*16+e (matches contiguous B).
__device__ __forceinline__ v16bf load_a_tile(const __bf16* tile, int row,
                                             int half)
{
    const v8bf lo = *(const v8bf*)&tile[row * 32 + half * 16];
    const v8bf hi = *(const v8bf*)&tile[row * 32 + half * 16 + 8];
    v16bf A;
    #pragma unroll
    for (int e = 0; e < 8; ++e) { A[e] = lo[e]; A[e + 8] = hi[e]; }
    return A;
}

// ---------------------------------------------------------------------------
// Kernel 1: Wh = h @ W in fp32 (protects the exp-sensitive f_src/f_dst path);
// emits bf16 Wh^T for WMMA B-operands. One block = 16 rows.
// ---------------------------------------------------------------------------
__global__ __launch_bounds__(256) void k_wh(const float* __restrict__ h,
                                            const float* __restrict__ Wm,
                                            const float* __restrict__ av,
                                            unsigned short* __restrict__ whbt_u,
                                            float* __restrict__ f_src,
                                            float* __restrict__ f_dst)
{
    __shared__ float sW[FIN * FOUT];   // 64 KB
    __shared__ float sh[16 * FIN];     // 16 KB
    __shared__ float sa[2 * FOUT];
    __shared__ float sred[256];
    __bf16* whbt = (__bf16*)whbt_u;

    const int tid  = threadIdx.x;
    const int row0 = blockIdx.x * 16;

    for (int i = tid; i < FIN * FOUT; i += 256) sW[i] = Wm[i];
    for (int i = tid; i < 16 * FIN;   i += 256) sh[i] = h[(size_t)row0 * FIN + i];
    if (tid < 2 * FOUT) sa[tid] = av[tid];
    __syncthreads();

    const int rl = tid >> 4;      // local row 0..15
    const int cg = tid & 15;      // cols 4cg..4cg+3
    const float* hr = &sh[rl * FIN];
    float acc0 = 0.f, acc1 = 0.f, acc2 = 0.f, acc3 = 0.f;
    #pragma unroll 4
    for (int k = 0; k < FIN; ++k) {
        const float hv = hr[k];
        const float* wr = &sW[k * FOUT + cg * 4];
        acc0 += hv * wr[0]; acc1 += hv * wr[1];
        acc2 += hv * wr[2]; acc3 += hv * wr[3];
    }
    const int row = row0 + rl;
    whbt[(size_t)(cg * 4 + 0) * NROW + row] = (__bf16)acc0;
    whbt[(size_t)(cg * 4 + 1) * NROW + row] = (__bf16)acc1;
    whbt[(size_t)(cg * 4 + 2) * NROW + row] = (__bf16)acc2;
    whbt[(size_t)(cg * 4 + 3) * NROW + row] = (__bf16)acc3;

    const float ps = acc0 * sa[cg*4+0] + acc1 * sa[cg*4+1]
                   + acc2 * sa[cg*4+2] + acc3 * sa[cg*4+3];
    const float pd = acc0 * sa[64+cg*4+0] + acc1 * sa[64+cg*4+1]
                   + acc2 * sa[64+cg*4+2] + acc3 * sa[64+cg*4+3];

    sred[tid] = ps; __syncthreads();
    if (cg == 0) {
        float s = 0.f;
        #pragma unroll
        for (int t = 0; t < 16; ++t) s += sred[rl * 16 + t];
        f_src[row] = s;
    }
    __syncthreads();
    sred[tid] = pd; __syncthreads();
    if (cg == 0) {
        float s = 0.f;
        #pragma unroll
        for (int t = 0; t < 16; ++t) s += sred[rl * 16 + t];
        f_dst[row] = s;
    }
}

// ---------------------------------------------------------------------------
// Kernel 2: M = max_j f_dst[j]  (softmax-shift bound; LeakyReLU monotone =>
// lrelu(f_src[i]+M) upper-bounds every score in row i).
// ---------------------------------------------------------------------------
__global__ __launch_bounds__(256) void k_max(const float* __restrict__ f_dst,
                                             float* __restrict__ Mmax)
{
    __shared__ float sm[256];
    const int tid = threadIdx.x;
    float m = -3.4e38f;
    for (int i = tid; i < NROW; i += 256) m = fmaxf(m, f_dst[i]);
    sm[tid] = m; __syncthreads();
    for (int s = 128; s > 0; s >>= 1) {
        if (tid < s) sm[tid] = fmaxf(sm[tid], sm[tid + s]);
        __syncthreads();
    }
    if (tid == 0) Mmax[0] = sm[0];
}

// ---------------------------------------------------------------------------
// Kernel 2b: per-column exp factors (separable LeakyReLU-softmax).
// ---------------------------------------------------------------------------
__global__ __launch_bounds__(256) void k_ecol(const float* __restrict__ f_dst,
                                              float* __restrict__ E1c,
                                              float* __restrict__ E2c)
{
    const int i = blockIdx.x * 256 + threadIdx.x;
    const float fd = f_dst[i];
    E1c[i] = __expf(fd);
    E2c[i] = __expf(ALPHA * fd);
}

// ---------------------------------------------------------------------------
// Kernel 3: the one HBM pass: adj (256 MiB fp32) -> 1-bit adjacency (8 MiB,
// L2-resident). Nontemporal loads; wave32 ballot packs 32 elems -> 1 word.
// ---------------------------------------------------------------------------
__global__ __launch_bounds__(256) void k_pack(const float* __restrict__ adj,
                                              unsigned int* __restrict__ mask)
{
    const int lane = threadIdx.x & 31;
    const int g    = blockIdx.x * 8 + (threadIdx.x >> 5);  // global wave id
    const int row  = g >> 3;
    const int seg  = g & 7;                                 // 1024-elem segment
    const float* p = adj + (size_t)row * NROW + seg * 1024;
    unsigned int* mw = mask + (size_t)row * NW + seg * 32;

    __builtin_prefetch(p + lane, 0, 0);   // global_prefetch_b8, speculative
    #pragma unroll 4
    for (int w = 0; w < 32; ++w) {
        const float v = __builtin_nontemporal_load(p + w * 32 + lane);
#if __has_builtin(__builtin_amdgcn_ballot_w32)
        const unsigned int b = __builtin_amdgcn_ballot_w32(v > 0.0f);
#else
        const unsigned int b = (unsigned int)__ballot(v > 0.0f);
#endif
        if (lane == 0) mw[w] = b;
    }
}

// ---------------------------------------------------------------------------
// Kernel 4: t = adj @ Wh. Block = 32 rows (2 M-tiles) x 64 cols, 4 waves
// (one 16-wide N-tile each). Mask block enters LDS via TDM; bf16 A tile built
// once per block via nibble LUT (double-buffered, 1 barrier/chunk); B is
// register-double-buffered so its wait lands an iteration after issue.
// ---------------------------------------------------------------------------
__global__ __launch_bounds__(128) void k_t(const unsigned int* __restrict__ mask,
                                           const unsigned short* __restrict__ whbt_u,
                                           unsigned short* __restrict__ tbt_u)
{
    __shared__ unsigned int smask[32 * NW];               // 32 KB
    __shared__ __align__(16) unsigned int sAbuf[2][512];  // 2 x 32x32 bf16
    __shared__ uint2 slut[16];                            // nibble -> 2 packed bf16
    const __bf16* whbt = (const __bf16*)whbt_u;
    __bf16* tbt = (__bf16*)tbt_u;

    const int tid  = threadIdx.x;
    const int wave = tid >> 5, lane = tid & 31;
    const int n    = lane & 15, half = lane >> 4;
    const int row0 = blockIdx.x * 32;

#if defined(HAVE_TDM)
    if (wave == 0) {
        tdm_load_mask(mask + (size_t)row0 * NW, smask, 32u);
        __builtin_amdgcn_s_wait_tensorcnt(0);
    }
#else
    for (int i = tid; i < 32 * NW; i += 128) smask[i] = mask[(size_t)row0 * NW + i];
#endif
    if (tid < 16) {
        const unsigned int b0 = tid & 1u, b1 = (tid >> 1) & 1u;
        const unsigned int b2 = (tid >> 2) & 1u, b3 = (tid >> 3) & 1u;
        uint2 e;
        e.x = b0 * 0x00003F80u + b1 * 0x3F800000u;
        e.y = b2 * 0x00003F80u + b3 * 0x3F800000u;
        slut[tid] = e;
    }
    __syncthreads();

    const int col = wave * 16 + n;
    const __bf16* bp = whbt + (size_t)col * NROW + half * 16;

    build_adj_tile(smask, slut, sAbuf[0], 0, tid);
    v16bf Bcur = *(const v16bf*)bp;
    __syncthreads();

    v8f acc0 = {0.f,0.f,0.f,0.f,0.f,0.f,0.f,0.f};
    v8f acc1 = {0.f,0.f,0.f,0.f,0.f,0.f,0.f,0.f};

    for (int c = 0; c < NW; ++c) {
        const int cur = c & 1;
        v16bf Bnext = Bcur;
        if (c + 1 < NW) {
            Bnext = *(const v16bf*)(bp + (size_t)(c + 1) * 32);        // prefetch
            build_adj_tile(smask, slut, sAbuf[cur ^ 1], c + 1, tid);   // next A
        }
        const __bf16* tile = (const __bf16*)sAbuf[cur];
        const v16bf A0 = load_a_tile(tile, n,      half);
        const v16bf A1 = load_a_tile(tile, 16 + n, half);
        acc0 = __builtin_amdgcn_wmma_f32_16x16x32_bf16(false, A0, false, Bcur,
                                                       (short)0, acc0, false, false);
        acc1 = __builtin_amdgcn_wmma_f32_16x16x32_bf16(false, A1, false, Bcur,
                                                       (short)0, acc1, false, false);
        __syncthreads();
        Bcur = Bnext;
    }

    // C/D: VGPR r -> row half*8 + r, col = lane&15 (+tile)
    #pragma unroll
    for (int r = 0; r < 8; ++r) {
        tbt[(size_t)col * NROW + row0 + half * 8 + r]      = (__bf16)acc0[r];
        tbt[(size_t)col * NROW + row0 + 16 + half * 8 + r] = (__bf16)acc1[r];
    }
    // augmentation: col 64 = ones (softmax denominator via WMMA), 65..79 = 0
    if (tid < 32) {
        const __bf16 one = (__bf16)1.0f, zero = (__bf16)0.0f;
        #pragma unroll
        for (int cc = 64; cc < 80; ++cc)
            tbt[(size_t)cc * NROW + row0 + tid] = (cc == 64) ? one : zero;
    }
}

// ---------------------------------------------------------------------------
// Kernel 5: one-pass masked softmax + h' = att @ [t | 1], ELU, fp32 out.
// Block = 32 rows, 5 waves: N-tiles 0..3 = numerator cols, tile 4 col 64 = Z.
// Exp-free inner loop (separable factors); P double-buffered, 1 barrier/chunk.
// ---------------------------------------------------------------------------
__global__ __launch_bounds__(160) void k_attn(const unsigned int* __restrict__ mask,
                                              const unsigned short* __restrict__ tbt_u,
                                              const float* __restrict__ f_src,
                                              const float* __restrict__ E1c,
                                              const float* __restrict__ E2c,
                                              const float* __restrict__ Mmax,
                                              float* __restrict__ out)
{
    __shared__ float sE1[NROW];                           // 32 KB exp(fd)
    __shared__ float sE2[NROW];                           // 32 KB exp(.2 fd)
    __shared__ unsigned int smask[32 * NW];               // 32 KB
    __shared__ __align__(16) unsigned int sPbuf[2][512];  // 2 x 32x32 bf16
    __shared__ float sR1[32], sR2[32], sTr[32], sZ[32];

    const __bf16* tbt = (const __bf16*)tbt_u;
    const int tid  = threadIdx.x;
    const int wave = tid >> 5, lane = tid & 31;
    const int n    = lane & 15, half = lane >> 4;
    const int row0 = blockIdx.x * 32;

#if defined(HAVE_TDM)
    if (wave == 0) {
        tdm_load_mask(mask + (size_t)row0 * NW, smask, 32u);
        __builtin_amdgcn_s_wait_tensorcnt(0);
    }
#else
    for (int i = tid; i < 32 * NW; i += 160) smask[i] = mask[(size_t)row0 * NW + i];
#endif
    for (int i = tid; i < NROW; i += 160) { sE1[i] = E1c[i]; sE2[i] = E2c[i]; }
    if (tid < 32) {
        const float fs = f_src[row0 + tid];
        const float t  = fs + Mmax[0];
        const float mi = (t > 0.f) ? t : ALPHA * t;   // row shift (upper bound)
        sR1[tid] = __expf(fs - mi);
        sR2[tid] = __expf(ALPHA * fs - mi);
        sTr[tid] = __expf(-fs);                       // branch threshold
    }
    __syncthreads();

    const int col = wave * 16 + n;
    const __bf16* bp = tbt + (size_t)col * NROW + half * 16;

    build_p_tile(smask, sE1, sE2, sR1, sR2, sTr, sPbuf[0], 0, tid);
    v16bf Bcur = *(const v16bf*)bp;
    __syncthreads();

    v8f acc0 = {0.f,0.f,0.f,0.f,0.f,0.f,0.f,0.f};
    v8f acc1 = {0.f,0.f,0.f,0.f,0.f,0.f,0.f,0.f};

    for (int c = 0; c < NW; ++c) {
        const int cur = c & 1;
        v16bf Bnext = Bcur;
        if (c + 1 < NW) {
            Bnext = *(const v16bf*)(bp + (size_t)(c + 1) * 32);        // prefetch
            build_p_tile(smask, sE1, sE2, sR1, sR2, sTr,
                         sPbuf[cur ^ 1], c + 1, tid);
        }
        const __bf16* tile = (const __bf16*)sPbuf[cur];
        const v16bf A0 = load_a_tile(tile, n,      half);
        const v16bf A1 = load_a_tile(tile, 16 + n, half);
        acc0 = __builtin_amdgcn_wmma_f32_16x16x32_bf16(false, A0, false, Bcur,
                                                       (short)0, acc0, false, false);
        acc1 = __builtin_amdgcn_wmma_f32_16x16x32_bf16(false, A1, false, Bcur,
                                                       (short)0, acc1, false, false);
        __syncthreads();
        Bcur = Bnext;
    }

    if (wave == 4 && n == 0) {                 // col 64 -> Z per row
        #pragma unroll
        for (int r = 0; r < 8; ++r) {
            sZ[half * 8 + r]      = acc0[r];
            sZ[16 + half * 8 + r] = acc1[r];
        }
    }
    __syncthreads();

    if (wave < 4) {
        #pragma unroll
        for (int r = 0; r < 8; ++r) {
            const int rl0 = half * 8 + r;
            const int rl1 = 16 + half * 8 + r;
            float v0 = acc0[r] / sZ[rl0];
            float v1 = acc1[r] / sZ[rl1];
            v0 = (v0 > 0.f) ? v0 : expm1f(v0);  // ELU
            v1 = (v1 > 0.f) ? v1 : expm1f(v1);
            out[(size_t)(row0 + rl0) * FOUT + col] = v0;
            out[(size_t)(row0 + rl1) * FOUT + col] = v1;
        }
    }
}

// ---------------------------------------------------------------------------
extern "C" void kernel_launch(void* const* d_in, const int* in_sizes, int n_in,
                              void* d_out, int out_size, void* d_ws, size_t ws_size,
                              hipStream_t stream)
{
    const float* h   = (const float*)d_in[0];   // [8192,256]
    const float* adj = (const float*)d_in[1];   // [8192,8192]
    const float* Wm  = (const float*)d_in[2];   // [256,64]
    const float* av  = (const float*)d_in[3];   // [128,1]
    float* out = (float*)d_out;                 // [8192,64] fp32
    char*  ws  = (char*)d_ws;

    unsigned short* whbt = (unsigned short*)(ws + WHBT_OFF);
    float* f_src = (float*)(ws + FSRC_OFF);
    float* f_dst = (float*)(ws + FDST_OFF);
    float* Mmax  = (float*)(ws + MMAX_OFF);
    unsigned int* mask = (unsigned int*)(ws + MASK_OFF);
    unsigned short* tbt = (unsigned short*)(ws + TBT_OFF);
    float* E1c = (float*)(ws + E1C_OFF);
    float* E2c = (float*)(ws + E2C_OFF);

    k_wh  <<<NROW / 16, 256, 0, stream>>>(h, Wm, av, whbt, f_src, f_dst);
    k_max <<<1,         256, 0, stream>>>(f_dst, Mmax);
    k_ecol<<<NROW / 256, 256, 0, stream>>>(f_dst, E1c, E2c);
    k_pack<<<NROW,      256, 0, stream>>>(adj, mask);
    k_t   <<<NROW / 32, 128, 0, stream>>>(mask, whbt, tbt);
    k_attn<<<NROW / 32, 160, 0, stream>>>(mask, tbt, f_src, E1c, E2c, Mmax, out);
}
